// SelfAttention_4346506903827
// MI455X (gfx1250) — matrix operand order repaired and checked by reading, not compiled, not run on hardware
//
#include <hip/hip_runtime.h>
#include <cstdint>
#include <cstddef>

#define DIM  1024
#define SEQ  2048
#define NB   16

typedef __attribute__((ext_vector_type(16))) __bf16 v16bf;
typedef __attribute__((ext_vector_type(8)))  __bf16 v8bf;
typedef __attribute__((ext_vector_type(8)))  float  v8f;
typedef __attribute__((ext_vector_type(4)))  float  v4f;
typedef __attribute__((ext_vector_type(2)))  float  v2f;

// ---------------------------------------------------------------------------
// WMMA fragment helpers (ISA 7.12.2 layouts, wave32)
// A 16x32 bf16: lane m(0-15): elems0-7 = K0-7, elems8-15 = K16-23;
//               lane m+16 :   elems0-7 = K8-15, elems8-15 = K24-31
// B 32x16 bf16: lane n(0-15) = column n, elems0-7 = K0-7, elems8-15 = K16-23; lane n+16: K8-15, K24-31 (as A)
// C/D 16x16 f32: vgpr r -> row r + 8*(lane>=16), col = lane&15
// ---------------------------------------------------------------------------

__device__ inline v16bf load_a_bf16(const __bf16* __restrict__ rowp, int k0, int hi) {
    const v8bf lo = *(const v8bf*)(rowp + k0 + hi * 8);
    const v8bf hh = *(const v8bf*)(rowp + k0 + 16 + hi * 8);
    v16bf r;
#pragma unroll
    for (int i = 0; i < 8; ++i) { r[i] = lo[i]; r[i + 8] = hh[i]; }
    return r;
}

// A frag built from f32 source with on-the-fly bf16 conversion
__device__ inline v16bf load_a_f32cvt(const float* __restrict__ rowp, int k0, int hi) {
    const v4f f0 = *(const v4f*)(rowp + k0 + hi * 8);
    const v4f f1 = *(const v4f*)(rowp + k0 + hi * 8 + 4);
    const v4f f2 = *(const v4f*)(rowp + k0 + 16 + hi * 8);
    const v4f f3 = *(const v4f*)(rowp + k0 + 16 + hi * 8 + 4);
    v16bf r;
#pragma unroll
    for (int i = 0; i < 4; ++i) {
        r[i]      = (__bf16)f0[i];
        r[i + 4]  = (__bf16)f1[i];
        r[i + 8]  = (__bf16)f2[i];
        r[i + 12] = (__bf16)f3[i];
    }
    return r;
}

// B frag: column `col` of B = row `col` of row-major matrix W[ld] (x@W^T form)
__device__ inline v16bf load_b_bf16(const __bf16* __restrict__ base, int ld,
                                    int col, int k0, int hi) {
    const __bf16* p = base + (size_t)col * ld + k0 + hi * 8;
    const v8bf lo = *(const v8bf*)p;
    const v8bf hh = *(const v8bf*)(p + 16);
    v16bf r;
#pragma unroll
    for (int i = 0; i < 8; ++i) { r[i] = lo[i]; r[i + 8] = hh[i]; }
    return r;
}

// B frag from an LDS-staged 16-row x 2048B key tile with 16B XOR swizzle:
// chunk at (row, byte_off) lives at row*2048 + (byte_off ^ ((row&15)<<4))
__device__ inline v16bf load_b_lds(const __bf16* buf, int row, int kchunk, int hi) {
    const int sw = (row & 15) << 4;
    const char* p = (const char*)buf + row * (DIM * 2);
    const int off0 = (kchunk + hi * 8) * 2;
    const v8bf lo = *(const v8bf*)(p + (off0 ^ sw));
    const v8bf hh = *(const v8bf*)(p + ((off0 + 32) ^ sw));
    v16bf r;
#pragma unroll
    for (int i = 0; i < 8; ++i) { r[i] = lo[i]; r[i + 8] = hh[i]; }
    return r;
}

__device__ inline v8f wmma_bf16(v16bf a, v16bf b, v8f c) {
    return __builtin_amdgcn_wmma_f32_16x16x32_bf16(false, a, false, b,
                                                   (short)0, c, false, false);
}

// Stage one 16-key x DIM bf16 tile (32KB) from global to LDS, swizzled.
// 8 waves split the tile (4KB each); per lane: 8 async 16B transfers.
__device__ inline void stage_ktile(const __bf16* krow0, __bf16* buf,
                                   int wave, int lane) {
#pragma unroll
    for (int i = 0; i < 8; ++i) {
        const int lin = wave * 4096 + i * 512 + lane * 16;  // byte off in 32KB tile
        const int row = lin >> 11;                          // /2048B per row
        const int off = lin & 2047;
        const int soff = off ^ ((row & 15) << 4);
        const uint32_t lds = (uint32_t)(uintptr_t)buf + row * (DIM * 2) + soff;
        const uint64_t ga  = (uint64_t)(uintptr_t)krow0 + (uint64_t)row * (DIM * 2) + off;
        asm volatile("global_load_async_to_lds_b128 %0, %1, off"
                     :: "v"(lds), "v"(ga) : "memory");
    }
}
__device__ inline void wait_async0() {
    asm volatile("s_wait_asynccnt 0x0" ::: "memory");
}

// ---------------------------------------------------------------------------
// K0: convert Wq/Wk to bf16
// ---------------------------------------------------------------------------
__global__ __launch_bounds__(256)
void cvtw_kernel(const float* __restrict__ Wq, const float* __restrict__ Wk,
                 unsigned short* __restrict__ Wqb, unsigned short* __restrict__ Wkb) {
    int i = blockIdx.x * 256 + threadIdx.x;
    if (i < DIM * DIM) {
        ((__bf16*)Wqb)[i] = (__bf16)Wq[i];
        ((__bf16*)Wkb)[i] = (__bf16)Wk[i];
    }
}

// ---------------------------------------------------------------------------
// K1: projection GEMM  Out(bf16) = x @ W^T + b      [32768 x 1024]
// ---------------------------------------------------------------------------
__global__ __launch_bounds__(256)
void proj_kernel(const float* __restrict__ x, const unsigned short* __restrict__ Wbf_,
                 const float* __restrict__ bias, unsigned short* __restrict__ Out_) {
    const __bf16* Wbf = (const __bf16*)Wbf_;
    __bf16* Out = (__bf16*)Out_;
    const int wave = threadIdx.x >> 5;
    const int lane = threadIdx.x & 31;
    const int lh = lane & 15, hi = lane >> 4;
    const int row0 = blockIdx.x * 128 + wave * 16;
    const int n0 = blockIdx.y * 128;
    const float* xrow = x + (size_t)(row0 + lh) * DIM;

    v8f acc[8];
#pragma unroll
    for (int t = 0; t < 8; ++t) acc[t] = {};

    for (int k0 = 0; k0 < DIM; k0 += 32) {
        v16bf afrag = load_a_f32cvt(xrow, k0, hi);
#pragma unroll
        for (int nt = 0; nt < 8; ++nt) {
            v16bf bfrag = load_b_bf16(Wbf, DIM, n0 + nt * 16 + lh, k0, hi);
            acc[nt] = wmma_bf16(afrag, bfrag, acc[nt]);
        }
    }
#pragma unroll
    for (int nt = 0; nt < 8; ++nt) {
        const int col = n0 + nt * 16 + lh;
        const float bv = bias[col];
#pragma unroll
        for (int r = 0; r < 8; ++r) {
            const int row = row0 + r + 8 * hi;
            Out[(size_t)row * DIM + col] = (__bf16)(acc[nt][r] + bv);
        }
    }
}

// ---------------------------------------------------------------------------
// K2: fused attention -> per-(batch,qblock,wave) attention column-sum partials
// grid (B=16, S/128=16); block 256 thr = 8 waves; wave owns 16 queries.
// Key tiles are async-staged to LDS (double buffered, 2x32KB, ASYNCcnt).
// Pass A: online softmax stats (m,l); Pass B: column sums of attn.
// Q A-fragments (32 chunks, ~256 VGPRs) stay register-resident.
// ---------------------------------------------------------------------------
__global__ __launch_bounds__(256)
void attn_colsum_kernel(const unsigned short* __restrict__ Q_,
                        const unsigned short* __restrict__ K_,
                        const int* __restrict__ mask,
                        float* __restrict__ wpart) {
    __shared__ __align__(16) __bf16 ktile[2][16 * DIM];   // 64 KB double buffer
    const __bf16* Q = (const __bf16*)Q_;
    const __bf16* Km = (const __bf16*)K_;

    const int b = blockIdx.x, qb = blockIdx.y;
    const int tid = threadIdx.x;
    const int wave = tid >> 5, lane = tid & 31;
    const int lh = lane & 15, hi = lane >> 4;

    const int q0 = qb * 128 + wave * 16;
    const __bf16* qrow = Q + ((size_t)(b * SEQ + q0) + lh) * DIM;
    const __bf16* kbase = Km + (size_t)b * SEQ * DIM;
    const int* mrow = mask + b * SEQ;
    const float scale = 0.03125f;             // 1/sqrt(1024)
    const int NKT = SEQ / 16;                 // 128 key tiles

    v16bf qa[32];
#pragma unroll
    for (int c = 0; c < 32; ++c) qa[c] = load_a_bf16(qrow, c * 32, hi);

    float m[8], l[8];
#pragma unroll
    for (int r = 0; r < 8; ++r) { m[r] = -INFINITY; l[r] = 0.f; }

    // ----- pass A: row max / sum-exp -----
    stage_ktile(kbase, ktile[0], wave, lane);
    for (int kt = 0; kt < NKT; ++kt) {
        const int cur = kt & 1;
        wait_async0();
        __syncthreads();                       // tile kt staged; buffers free
        if (kt + 1 < NKT)
            stage_ktile(kbase + (size_t)(kt + 1) * 16 * DIM, ktile[1 - cur], wave, lane);

        const int k0 = kt * 16;
        const bool masked = (mrow[k0 + lh] == 0);
        v8f acc = {};
#pragma unroll
        for (int c = 0; c < 32; ++c)
            acc = wmma_bf16(qa[c], load_b_lds(ktile[cur], lh, c * 32, hi), acc);
#pragma unroll
        for (int r = 0; r < 8; ++r) {
            const float s = acc[r] * scale;
            float v = masked ? -INFINITY : s;
#pragma unroll
            for (int off = 8; off; off >>= 1) v = fmaxf(v, __shfl_xor(v, off, 16));
            const float mnew = fmaxf(m[r], v);
            float term = masked ? 0.f : __expf(s - mnew);
#pragma unroll
            for (int off = 8; off; off >>= 1) term += __shfl_xor(term, off, 16);
            const float corr = (m[r] == mnew) ? 1.f : __expf(m[r] - mnew);
            l[r] = l[r] * corr + term;
            m[r] = mnew;
        }
    }
    float rl[8];
#pragma unroll
    for (int r = 0; r < 8; ++r) rl[r] = (l[r] > 0.f) ? 1.f / l[r] : 0.f;

    // ----- pass B: column sums of normalized attention -----
    float* wp = wpart + (size_t)((b * 16 + qb) * 8 + wave) * SEQ;
    stage_ktile(kbase, ktile[0], wave, lane);
    for (int kt = 0; kt < NKT; ++kt) {
        const int cur = kt & 1;
        wait_async0();
        __syncthreads();
        if (kt + 1 < NKT)
            stage_ktile(kbase + (size_t)(kt + 1) * 16 * DIM, ktile[1 - cur], wave, lane);

        const int k0 = kt * 16;
        const bool masked = (mrow[k0 + lh] == 0);
        v8f acc = {};
#pragma unroll
        for (int c = 0; c < 32; ++c)
            acc = wmma_bf16(qa[c], load_b_lds(ktile[cur], lh, c * 32, hi), acc);
        float cs = 0.f;
#pragma unroll
        for (int r = 0; r < 8; ++r) {
            const float s = acc[r] * scale;
            const float p = masked ? 0.f : __expf(s - m[r]) * rl[r];
            cs += p;
        }
        cs += __shfl_xor(cs, 16, 32);          // combine row-halves of the column
        if (hi == 0) wp[k0 + lh] = cs;         // each (wave,key) written once
    }
}

// ---------------------------------------------------------------------------
// K3: w[b,k] = sum over 16 query-blocks x 8 waves of wpart (fixed order)
// ---------------------------------------------------------------------------
__global__ __launch_bounds__(256)
void reduce_w_kernel(const float* __restrict__ wpart, float* __restrict__ w) {
    int i = blockIdx.x * 256 + threadIdx.x;
    if (i < NB * SEQ) {
        const int b = i / SEQ, k = i % SEQ;
        float s = 0.f;
        for (int j = 0; j < 128; ++j) s += wpart[(size_t)(b * 128 + j) * SEQ + k];
        w[i] = s;
    }
}

// ---------------------------------------------------------------------------
// K4a: y[b,d] = sum_k w[b,k] * x[b,k,d]
// ---------------------------------------------------------------------------
__global__ __launch_bounds__(256)
void wx_kernel(const float* __restrict__ x, const float* __restrict__ w,
               float* __restrict__ y) {
    const int b = blockIdx.y;
    const int d = blockIdx.x * 256 + threadIdx.x;
    const float* xb = x + (size_t)b * SEQ * DIM + d;
    const float* wb = w + b * SEQ;
    float acc = 0.f;
    for (int k = 0; k < SEQ; ++k) acc += wb[k] * xb[(size_t)k * DIM];
    y[b * DIM + d] = acc;
}

// K4b: wsum[b] = sum_k w[b,k] (deterministic tree)
__global__ __launch_bounds__(256)
void wsum_kernel(const float* __restrict__ w, float* __restrict__ wsum) {
    __shared__ float red[256];
    const int b = blockIdx.x, t = threadIdx.x;
    float s = 0.f;
    for (int k = t; k < SEQ; k += 256) s += w[b * SEQ + k];
    red[t] = s; __syncthreads();
    for (int st = 128; st > 0; st >>= 1) {
        if (t < st) red[t] += red[t + st];
        __syncthreads();
    }
    if (t == 0) wsum[b] = red[0];
}

// ---------------------------------------------------------------------------
// K5: out[b,e] = ( y[b,:]·Wv[e,:] + wsum[b]*bv[e] ) / S   via f32 WMMA 16x16x4
// ---------------------------------------------------------------------------
__global__ __launch_bounds__(32)
void final_kernel(const float* __restrict__ y, const float* __restrict__ Wv,
                  const float* __restrict__ bv, const float* __restrict__ wsum,
                  float* __restrict__ out) {
    const int lane = threadIdx.x & 31;
    const int lh = lane & 15, hi = lane >> 4;
    const int n0 = blockIdx.x * 16;
    v8f acc = {};
    const float* yrow = y + (size_t)lh * DIM;             // A row = batch lh
    const float* wrow = Wv + (size_t)(n0 + lh) * DIM;     // B col = Wv row
    for (int k0 = 0; k0 < DIM; k0 += 4) {
        v2f a = *(const v2f*)(yrow + k0 + hi * 2);        // K = hi*2 + {0,1}
        v2f bb = *(const v2f*)(wrow + k0 + hi * 2);
        acc = __builtin_amdgcn_wmma_f32_16x16x4_f32(false, a, false, bb,
                                                    (short)0, acc, false, false);
    }
    const int col = n0 + lh;
    const float bvv = bv[col];
#pragma unroll
    for (int r = 0; r < 8; ++r) {
        const int row = r + 8 * hi;                       // batch index
        out[row * DIM + col] = (acc[r] + wsum[row] * bvv) * (1.0f / (float)SEQ);
    }
}

// ---------------------------------------------------------------------------
extern "C" void kernel_launch(void* const* d_in, const int* in_sizes, int n_in,
                              void* d_out, int out_size, void* d_ws, size_t ws_size,
                              hipStream_t stream) {
    const float* x    = (const float*)d_in[0];
    const int*   mask = (const int*)d_in[1];
    const float* Wq   = (const float*)d_in[2];
    const float* bq   = (const float*)d_in[3];
    const float* Wk   = (const float*)d_in[4];
    const float* bk   = (const float*)d_in[5];
    const float* Wv   = (const float*)d_in[6];
    const float* bv   = (const float*)d_in[7];
    float* out = (float*)d_out;

    char* ws = (char*)d_ws;
    const size_t ROWS   = (size_t)NB * SEQ;                    // 32768
    const size_t offWqb = 0;                                   // 2 MB
    const size_t offWkb = offWqb + (size_t)DIM * DIM * 2;      // 2 MB
    const size_t offQb  = offWkb + (size_t)DIM * DIM * 2;      // 64 MB
    const size_t offKb  = offQb + ROWS * DIM * 2;              // 64 MB
    const size_t offWp  = offKb + ROWS * DIM * 2;              // 16 MB (16*128*2048*4)
    const size_t offW   = offWp + (size_t)NB * 128 * SEQ * 4;  // 128 KB
    const size_t offY   = offW + (size_t)NB * SEQ * 4;         // 64 KB
    const size_t offWs  = offY + (size_t)NB * DIM * 4;

    unsigned short* Wqb = (unsigned short*)(ws + offWqb);
    unsigned short* Wkb = (unsigned short*)(ws + offWkb);
    unsigned short* Qb  = (unsigned short*)(ws + offQb);
    unsigned short* Kb  = (unsigned short*)(ws + offKb);
    float* wpart = (float*)(ws + offWp);
    float* w     = (float*)(ws + offW);
    float* y     = (float*)(ws + offY);
    float* wsum  = (float*)(ws + offWs);

    // 1) weights -> bf16
    cvtw_kernel<<<dim3((DIM * DIM + 255) / 256), 256, 0, stream>>>(Wq, Wk, Wqb, Wkb);
    // 2) Q, K projections
    proj_kernel<<<dim3(ROWS / 128, DIM / 128), 256, 0, stream>>>(x, Wqb, bq, Qb);
    proj_kernel<<<dim3(ROWS / 128, DIM / 128), 256, 0, stream>>>(x, Wkb, bk, Kb);
    // 3) fused attention -> attention column-sum partials (async LDS staging)
    attn_colsum_kernel<<<dim3(NB, SEQ / 128), 256, 0, stream>>>(Qb, Kb, mask, wpart);
    // 4) reduce partials, weighted x rows, total weight
    reduce_w_kernel<<<dim3((NB * SEQ + 255) / 256), 256, 0, stream>>>(wpart, w);
    wx_kernel<<<dim3(DIM / 256, NB), 256, 0, stream>>>(x, w, y);
    wsum_kernel<<<dim3(NB), 256, 0, stream>>>(w, wsum);
    // 5) final small GEMM (f32 WMMA) + bias + 1/S
    final_kernel<<<dim3(DIM / 16), 32, 0, stream>>>(y, Wv, bv, wsum, out);
}
